// PathInteractionGNN_49778670961189
// MI455X (gfx1250) — matrix operand order repaired and checked by reading, not compiled, not run on hardware
//
#include <hip/hip_runtime.h>
#include <hip/hip_bf16.h>

#define N_NODES 50000
#define EMB     128
#define HID     64
#define HEADS   3
#define HC      (HEADS * HID)   // 192
#define NEDGE   800000

typedef __attribute__((ext_vector_type(16))) __bf16 v16bf;
typedef __attribute__((ext_vector_type(8)))  float  v8f;

// ---------------------------------------------------------------------------
// Workspace init: amax = -inf, asum = 0, outacc = 0
// ---------------------------------------------------------------------------
__global__ void init_ws_kernel(float* __restrict__ amax, float* __restrict__ asum,
                               float* __restrict__ outacc) {
    int i = blockIdx.x * blockDim.x + threadIdx.x;
    if (i < N_NODES * HEADS) {
        amax[i] = -3.0e38f;
        asum[i] = 0.0f;
    }
    if (i < N_NODES * HC) {
        outacc[i] = 0.0f;
    }
}

// ---------------------------------------------------------------------------
// Fused dual GEMM on the matrix pipe:
//   xl = path_emb @ Wl.T + bl ; xr = path_emb @ Wr.T + br
// A tile (path_emb rows, fp32 -> bf16) is loaded once per K-step and reused
// for two v_wmma_f32_16x16x32_bf16 ops. 3125 M-tiles x 12 N-tiles, K = 4x32.
// blockDim = 128 (4 waves); grid = (3125, 3); wave w handles jTile = by*4+w.
// All lanes always active (WMMA requires EXEC all-ones).
// ---------------------------------------------------------------------------
__global__ void gemm_xlxr_wmma(const float* __restrict__ X,
                               const float* __restrict__ Wl, const float* __restrict__ bl,
                               const float* __restrict__ Wr, const float* __restrict__ br,
                               float* __restrict__ xl, float* __restrict__ xr) {
    const int lane  = threadIdx.x & 31;
    const int wave  = threadIdx.x >> 5;                 // 0..3
    const int mTile = blockIdx.x;                       // 0..3124
    const int jTile = blockIdx.y * 4 + wave;            // 0..11
    const int row   = lane & 15;
    const int half  = lane >> 4;                        // 0 or 1
    const int m0    = mTile * 16;
    const int j0    = jTile * 16;

    v8f accL = {};
    v8f accR = {};

    #pragma unroll
    for (int kc = 0; kc < 4; ++kc) {
        const int k0 = kc * 32;

        // A fragment: 16x32 bf16.  Lane half 0: K = 0..15;
        // lane half 1: K = 16..31.  Row M = lane & 15.
        v16bf a;
        const float* arow = X + (size_t)(m0 + row) * EMB + k0 + half * 16;
        #pragma unroll
        for (int i = 0; i < 8; ++i) {
            a[i]     = (__bf16)arow[i];
            a[i + 8] = (__bf16)arow[8 + i];
        }

        // B fragment: 32x16 bf16, B[k][n] = W[j0+n][k].
        // Lanes 0-15 hold K=0..15, lanes 16-31 hold K=16..31; column = lane&15.
        v16bf bL, bR;
        const int   col   = row;
        const int   kb    = k0 + half * 16;
        const float* wlr  = Wl + (size_t)(j0 + col) * EMB + kb;
        const float* wrr  = Wr + (size_t)(j0 + col) * EMB + kb;
        #pragma unroll
        for (int i = 0; i < 16; ++i) {
            bL[i] = (__bf16)wlr[i];
            bR[i] = (__bf16)wrr[i];
        }

        accL = __builtin_amdgcn_wmma_f32_16x16x32_bf16(
            false, a, false, bL, (short)0, accL, false, false);
        accR = __builtin_amdgcn_wmma_f32_16x16x32_bf16(
            false, a, false, bR, (short)0, accR, false, false);
    }

    // D layout (16x16 f32): M = half*8 + r, N = lane & 15.
    const int n  = lane & 15;
    const float biasL = bl[j0 + n];
    const float biasR = br[j0 + n];
    #pragma unroll
    for (int r = 0; r < 8; ++r) {
        const int M   = half * 8 + r;
        const size_t idx = (size_t)(m0 + M) * HC + j0 + n;
        xl[idx] = accL[r] + biasL;
        xr[idx] = accR[r] + biasR;
    }
}

// ---------------------------------------------------------------------------
// Float atomic max via CAS
// ---------------------------------------------------------------------------
__device__ __forceinline__ void atomicMaxF(float* addr, float value) {
    int* iaddr = (int*)addr;
    int  old   = __float_as_int(*addr);
    while (value > __int_as_float(old)) {
        int assumed = old;
        old = atomicCAS(iaddr, assumed, __float_as_int(value));
        if (old == assumed) break;
    }
}

// ---------------------------------------------------------------------------
// Pass 1: attention logits per (edge, head); segment max via atomics.
//   a[e,h] = sum_c leaky_relu(xl[src,h,c] + xr[dst,h,c] + ea[e]*We[h,c]) * att[h,c]
// (softmax over the single sim_w channel is exactly 1 -> fused = edge_attr)
// Gathers are explicit b128 (rows are 256B-aligned); per-edge streaming
// traffic is non-temporal so it doesn't evict the L2-resident xl/xr.
// ---------------------------------------------------------------------------
__global__ void edge_logits_kernel(const int* __restrict__ ei,
                                   const float* __restrict__ ea,
                                   const float* __restrict__ xl,
                                   const float* __restrict__ xr,
                                   const float* __restrict__ We,
                                   const float* __restrict__ att,
                                   float* __restrict__ a_out,
                                   float* __restrict__ amax) {
    int t = blockIdx.x * blockDim.x + threadIdx.x;
    if (t >= NEDGE * HEADS) return;
    const int e = t / HEADS;
    const int h = t - e * HEADS;
    const int src = ei[e];
    const int dst = ei[NEDGE + e];
    const float f = __builtin_nontemporal_load(&ea[e]);

    const float4* xs4 = (const float4*)(xl  + (size_t)src * HC + h * HID);
    const float4* xd4 = (const float4*)(xr  + (size_t)dst * HC + h * HID);
    const float4* we4 = (const float4*)(We  + h * HID);
    const float4* at4 = (const float4*)(att + h * HID);
    __builtin_prefetch(xs4, 0, 0);
    __builtin_prefetch(xd4, 0, 0);

    float s = 0.0f;
    #pragma unroll
    for (int q = 0; q < HID / 4; ++q) {
        const float4 vs = xs4[q];
        const float4 vd = xd4[q];
        const float4 vw = we4[q];
        const float4 va = at4[q];
        float m0 = vs.x + vd.x + f * vw.x;
        float m1 = vs.y + vd.y + f * vw.y;
        float m2 = vs.z + vd.z + f * vw.z;
        float m3 = vs.w + vd.w + f * vw.w;
        m0 = m0 > 0.0f ? m0 : 0.2f * m0;
        m1 = m1 > 0.0f ? m1 : 0.2f * m1;
        m2 = m2 > 0.0f ? m2 : 0.2f * m2;
        m3 = m3 > 0.0f ? m3 : 0.2f * m3;
        s += m0 * va.x + m1 * va.y + m2 * va.z + m3 * va.w;
    }
    __builtin_nontemporal_store(s, &a_out[t]);
    atomicMaxF(&amax[dst * HEADS + h], s);
}

// ---------------------------------------------------------------------------
// Pass 2: aexp = exp(a - amax[dst]); segment sum. (overwrites a in place)
// ---------------------------------------------------------------------------
__global__ void edge_exp_kernel(const int* __restrict__ ei,
                                float* __restrict__ a_buf,
                                const float* __restrict__ amax,
                                float* __restrict__ asum) {
    int t = blockIdx.x * blockDim.x + threadIdx.x;
    if (t >= NEDGE * HEADS) return;
    const int e = t / HEADS;
    const int h = t - e * HEADS;
    const int dst = ei[NEDGE + e];
    const float logit = __builtin_nontemporal_load(&a_buf[t]);
    const float aexp  = __expf(logit - amax[dst * HEADS + h]);
    __builtin_nontemporal_store(aexp, &a_buf[t]);
    atomicAdd(&asum[dst * HEADS + h], aexp);
}

// ---------------------------------------------------------------------------
// Pass 3: alpha = aexp/asum[dst]; scatter-add alpha * xl[src] into outacc[dst].
// ---------------------------------------------------------------------------
__global__ void edge_scatter_kernel(const int* __restrict__ ei,
                                    const float* __restrict__ aexp,
                                    const float* __restrict__ asum,
                                    const float* __restrict__ xl,
                                    float* __restrict__ outacc) {
    int t = blockIdx.x * blockDim.x + threadIdx.x;
    if (t >= NEDGE * HEADS) return;
    const int e = t / HEADS;
    const int h = t - e * HEADS;
    const int src = ei[e];
    const int dst = ei[NEDGE + e];
    const float alpha = __builtin_nontemporal_load(&aexp[t]) / asum[dst * HEADS + h];

    const float4* xs4 = (const float4*)(xl + (size_t)src * HC + h * HID);
    float*        od  = outacc + (size_t)dst * HC + h * HID;
    __builtin_prefetch(xs4, 0, 0);
    #pragma unroll
    for (int q = 0; q < HID / 4; ++q) {
        const float4 v = xs4[q];
        atomicAdd(&od[4 * q + 0], alpha * v.x);
        atomicAdd(&od[4 * q + 1], alpha * v.y);
        atomicAdd(&od[4 * q + 2], alpha * v.z);
        atomicAdd(&od[4 * q + 3], alpha * v.w);
    }
}

// ---------------------------------------------------------------------------
// Pass 4: head-mean + bias, project to scalar score per node. (b128 loads)
// ---------------------------------------------------------------------------
__global__ void node_scores_kernel(const float* __restrict__ outacc,
                                   const float* __restrict__ bias_gnn,
                                   const float* __restrict__ Wout,
                                   const float* __restrict__ bout,
                                   float* __restrict__ scores) {
    int n = blockIdx.x * blockDim.x + threadIdx.x;
    if (n >= N_NODES) return;
    const float4* p0 = (const float4*)(outacc + (size_t)n * HC);
    const float4* p1 = (const float4*)(outacc + (size_t)n * HC + HID);
    const float4* p2 = (const float4*)(outacc + (size_t)n * HC + 2 * HID);
    const float4* bg = (const float4*)bias_gnn;
    const float4* wo = (const float4*)Wout;
    float s = 0.0f;
    #pragma unroll
    for (int q = 0; q < HID / 4; ++q) {
        const float4 a = p0[q];
        const float4 b = p1[q];
        const float4 c = p2[q];
        const float4 g = bg[q];
        const float4 w = wo[q];
        s += ((a.x + b.x + c.x) * (1.0f / 3.0f) + g.x) * w.x;
        s += ((a.y + b.y + c.y) * (1.0f / 3.0f) + g.y) * w.y;
        s += ((a.z + b.z + c.z) * (1.0f / 3.0f) + g.z) * w.z;
        s += ((a.w + b.w + c.w) * (1.0f / 3.0f) + g.w) * w.w;
    }
    scores[n] = s + bout[0];
}

// ---------------------------------------------------------------------------
// Global softmax over 50000 scores: partial max -> final max -> partial
// exp-sum -> final sum -> normalize.
// ---------------------------------------------------------------------------
#define RED_BLOCKS 256
#define RED_TPB    256

__global__ void reduce_max_partial(const float* __restrict__ scores,
                                   float* __restrict__ part) {
    __shared__ float sm[RED_TPB];
    float m = -3.0e38f;
    for (int i = blockIdx.x * RED_TPB + threadIdx.x; i < N_NODES;
         i += RED_BLOCKS * RED_TPB) {
        m = fmaxf(m, scores[i]);
    }
    sm[threadIdx.x] = m;
    __syncthreads();
    for (int s = RED_TPB / 2; s > 0; s >>= 1) {
        if (threadIdx.x < s) sm[threadIdx.x] = fmaxf(sm[threadIdx.x], sm[threadIdx.x + s]);
        __syncthreads();
    }
    if (threadIdx.x == 0) part[blockIdx.x] = sm[0];
}

__global__ void reduce_max_final(const float* __restrict__ part,
                                 float* __restrict__ gred) {
    __shared__ float sm[RED_TPB];
    sm[threadIdx.x] = (threadIdx.x < RED_BLOCKS) ? part[threadIdx.x] : -3.0e38f;
    __syncthreads();
    for (int s = RED_TPB / 2; s > 0; s >>= 1) {
        if (threadIdx.x < s) sm[threadIdx.x] = fmaxf(sm[threadIdx.x], sm[threadIdx.x + s]);
        __syncthreads();
    }
    if (threadIdx.x == 0) gred[0] = sm[0];
}

__global__ void reduce_sum_partial(const float* __restrict__ scores,
                                   const float* __restrict__ gred,
                                   float* __restrict__ part) {
    __shared__ float sm[RED_TPB];
    const float gmax = gred[0];
    float s = 0.0f;
    for (int i = blockIdx.x * RED_TPB + threadIdx.x; i < N_NODES;
         i += RED_BLOCKS * RED_TPB) {
        s += __expf(scores[i] - gmax);
    }
    sm[threadIdx.x] = s;
    __syncthreads();
    for (int st = RED_TPB / 2; st > 0; st >>= 1) {
        if (threadIdx.x < st) sm[threadIdx.x] += sm[threadIdx.x + st];
        __syncthreads();
    }
    if (threadIdx.x == 0) part[blockIdx.x] = sm[0];
}

__global__ void reduce_sum_final(const float* __restrict__ part,
                                 float* __restrict__ gred) {
    __shared__ float sm[RED_TPB];
    sm[threadIdx.x] = (threadIdx.x < RED_BLOCKS) ? part[threadIdx.x] : 0.0f;
    __syncthreads();
    for (int s = RED_TPB / 2; s > 0; s >>= 1) {
        if (threadIdx.x < s) sm[threadIdx.x] += sm[threadIdx.x + s];
        __syncthreads();
    }
    if (threadIdx.x == 0) gred[1] = sm[0];
}

__global__ void softmax_normalize(const float* __restrict__ scores,
                                  const float* __restrict__ gred,
                                  float* __restrict__ out) {
    int n = blockIdx.x * blockDim.x + threadIdx.x;
    if (n >= N_NODES) return;
    out[n] = __expf(scores[n] - gred[0]) / gred[1];
}

// ---------------------------------------------------------------------------
// Launch
// ---------------------------------------------------------------------------
extern "C" void kernel_launch(void* const* d_in, const int* in_sizes, int n_in,
                              void* d_out, int out_size, void* d_ws, size_t ws_size,
                              hipStream_t stream) {
    (void)in_sizes; (void)n_in; (void)out_size; (void)ws_size;

    const int*   edge_index = (const int*)  d_in[0];   // [2, E]
    const float* edge_attr  = (const float*)d_in[1];   // [E, 1]
    const float* path_emb   = (const float*)d_in[2];   // [N, 128]
    // d_in[3] = sim_w: softmax over a single channel == 1.0, unused.
    const float* Wl         = (const float*)d_in[4];   // [192, 128]
    const float* bl         = (const float*)d_in[5];   // [192]
    const float* Wr         = (const float*)d_in[6];   // [192, 128]
    const float* br         = (const float*)d_in[7];   // [192]
    const float* We         = (const float*)d_in[8];   // [192, 1]
    const float* att        = (const float*)d_in[9];   // [1, 3, 64]
    const float* bias_gnn   = (const float*)d_in[10];  // [64]
    const float* Wout       = (const float*)d_in[11];  // [1, 64]
    const float* bout       = (const float*)d_in[12];  // [1]
    float*       out        = (float*)d_out;           // [N]

    // Workspace layout (floats)
    float* ws     = (float*)d_ws;
    float* xl     = ws;                            // N*HC   = 9,600,000
    float* xr     = xl   + (size_t)N_NODES * HC;
    float* a_buf  = xr   + (size_t)N_NODES * HC;   // E*HEADS = 2,400,000
    float* amax   = a_buf + (size_t)NEDGE * HEADS; // N*HEADS
    float* asum   = amax + (size_t)N_NODES * HEADS;
    float* outacc = asum + (size_t)N_NODES * HEADS; // N*HC
    float* scores = outacc + (size_t)N_NODES * HC;  // N
    float* part   = scores + N_NODES;               // RED_BLOCKS
    float* gred   = part + RED_BLOCKS;              // 2: gmax, gsum

    // 0) re-init accumulators every call (deterministic, capture-safe)
    {
        int total = N_NODES * HC;
        init_ws_kernel<<<(total + 255) / 256, 256, 0, stream>>>(amax, asum, outacc);
    }

    // 1) dual WMMA GEMM: xl, xr
    {
        dim3 grid(N_NODES / 16, HC / (16 * 4));   // (3125, 3)
        gemm_xlxr_wmma<<<grid, 128, 0, stream>>>(path_emb, Wl, bl, Wr, br, xl, xr);
    }

    const int eh_threads = NEDGE * HEADS;
    const int eh_blocks  = (eh_threads + 255) / 256;

    // 2) attention logits + segment max
    edge_logits_kernel<<<eh_blocks, 256, 0, stream>>>(edge_index, edge_attr, xl, xr,
                                                      We, att, a_buf, amax);
    // 3) exp + segment sum
    edge_exp_kernel<<<eh_blocks, 256, 0, stream>>>(edge_index, a_buf, amax, asum);
    // 4) weighted scatter of source features
    edge_scatter_kernel<<<eh_blocks, 256, 0, stream>>>(edge_index, a_buf, asum, xl, outacc);
    // 5) head-mean + bias + output projection
    node_scores_kernel<<<(N_NODES + 255) / 256, 256, 0, stream>>>(outacc, bias_gnn,
                                                                  Wout, bout, scores);
    // 6) global softmax
    reduce_max_partial<<<RED_BLOCKS, RED_TPB, 0, stream>>>(scores, part);
    reduce_max_final<<<1, RED_TPB, 0, stream>>>(part, gred);
    reduce_sum_partial<<<RED_BLOCKS, RED_TPB, 0, stream>>>(scores, gred, part);
    reduce_sum_final<<<1, RED_TPB, 0, stream>>>(part, gred);
    softmax_normalize<<<(N_NODES + 255) / 256, 256, 0, stream>>>(scores, gred, out);
}
